// CTBlock_83786222010440
// MI455X (gfx1250) — matrix operand order repaired and checked by reading, not compiled, not run on hardware
//
#include <hip/hip_runtime.h>

// ---------------------------------------------------------------------------
// CDNA5 (gfx1250) implementation of CTBlock.
// All GEMM-like work (convs via implicit GEMM, linears, attention, LRED
// einsums) runs on V_WMMA_F32_16X16X32_F16 (f16 multiplicands staged through
// LDS, f32 accumulate). Staging is branch-free and vectorized; guard handling
// exploits that every K/Cin here is a multiple of 16 (all-or-nothing runs).
// Convs stage a full tap (all Cin) per barrier pair: Cin/32 WMMAs per sync.
// ---------------------------------------------------------------------------

typedef __attribute__((ext_vector_type(16))) _Float16 v16h;
typedef __attribute__((ext_vector_type(8)))  _Float16 v8h;
typedef __attribute__((ext_vector_type(8)))  float    v8f;
typedef __attribute__((ext_vector_type(4)))  float    f32x4;

#define CONV_CMAX 192              // max Cin over all convs in this net
#define CONV_LD   (CONV_CMAX + 8)  // 200 halfs -> 400B rows (16B aligned)

static __device__ __forceinline__ v8f wmma16(v16h a, v16h b, v8f c) {
  return __builtin_amdgcn_wmma_f32_16x16x32_f16(false, a, false, b, (short)0, c,
                                                false, false);
}

// A fragment from m-major tile (ld halfs/row, 16B-aligned rows):
// lane: m = L%16, g = L/16; elem e -> K=16g+e
static __device__ __forceinline__ v16h frag_a_ld(const _Float16* tile, int ld) {
  int lane = threadIdx.x & 31;
  int m = lane & 15, g = lane >> 4;
  const _Float16* r = tile + m * ld;
  v8h lo = *(const v8h*)(r + 16 * g);
  v8h hi = *(const v8h*)(r + 16 * g + 8);
  return __builtin_shufflevector(lo, hi, 0,1,2,3,4,5,6,7,8,9,10,11,12,13,14,15);
}

// B fragment from n-major tile: lane n = L%16, g = L/16; elem e -> K=16g+e
// -> 16 contiguous halfs per lane (two b128 DS loads).
static __device__ __forceinline__ v16h frag_b_ld(const _Float16* tile, int ld) {
  int lane = threadIdx.x & 31;
  int n = lane & 15, g = lane >> 4;
  const _Float16* r = tile + n * ld + 16 * g;
  v8h lo = *(const v8h*)(r);
  v8h hi = *(const v8h*)(r + 8);
  return __builtin_shufflevector(lo, hi, 0,1,2,3,4,5,6,7,8,9,10,11,12,13,14,15);
}

// 16 floats (4x float4) -> 16 contiguous f16 in LDS
static __device__ __forceinline__ void cvt_store16(_Float16* dst, f32x4 u0,
                                                   f32x4 u1, f32x4 u2, f32x4 u3) {
  v8h h0, h1;
#pragma unroll
  for (int e = 0; e < 4; ++e) {
    h0[e] = (_Float16)u0[e]; h0[4 + e] = (_Float16)u1[e];
    h1[e] = (_Float16)u2[e]; h1[4 + e] = (_Float16)u3[e];
  }
  *(v8h*)dst = h0; *(v8h*)(dst + 8) = h1;
}

// 16 floats -> scatter as column (stride ld rows)
static __device__ __forceinline__ void cvt_scatter16(_Float16* dst, int ld,
                                                     f32x4 u0, f32x4 u1,
                                                     f32x4 u2, f32x4 u3) {
#pragma unroll
  for (int e = 0; e < 4; ++e) {
    dst[e * ld]        = (_Float16)u0[e];
    dst[(4 + e) * ld]  = (_Float16)u1[e];
    dst[(8 + e) * ld]  = (_Float16)u2[e];
    dst[(12 + e) * ld] = (_Float16)u3[e];
  }
}

// 16 strided scalar loads -> contiguous LDS
static __device__ __forceinline__ void ldstrided_store16(_Float16* dst,
                                                         const float* p, long stride) {
  v8h h0, h1;
#pragma unroll
  for (int e = 0; e < 8; ++e) h0[e] = (_Float16)p[(long)e * stride];
#pragma unroll
  for (int e = 0; e < 8; ++e) h1[e] = (_Float16)p[(long)(8 + e) * stride];
  *(v8h*)dst = h0; *(v8h*)(dst + 8) = h1;
}

// All-or-nothing masked run: unconditional loads from a safe (clamped) base
// with a pre-selected stride (0 when the run is invalid), single select-zero.
static __device__ __forceinline__ void ldmask_store16(_Float16* dst,
                                                      const float* p, long stride,
                                                      bool ok) {
  v8h h0, h1;
#pragma unroll
  for (int e = 0; e < 8; ++e) {
    float v = p[(long)e * stride];
    h0[e] = (_Float16)(ok ? v : 0.f);
  }
#pragma unroll
  for (int e = 0; e < 8; ++e) {
    float v = p[(long)(8 + e) * stride];
    h1[e] = (_Float16)(ok ? v : 0.f);
  }
  *(v8h*)dst = h0; *(v8h*)(dst + 8) = h1;
}

// ---------------------------------------------------------------------------
// Generic batched GEMM:  C[m,n] = sum_k A[m,k]*B[k,n]  (+bias[m]) (+res) (act)
// Strided operands, two-level batch. Residual shares C addressing (own batch
// stride) which also realizes the torch ".view" flat adds.
// Requires: K a multiple of 16 for correct tail masking (true here).
// ---------------------------------------------------------------------------
__global__ void __launch_bounds__(128)
k_gemm(const float* __restrict__ A, const float* __restrict__ B,
       const float* __restrict__ bias, const float* __restrict__ res,
       float* __restrict__ C, int M, int N, int K,
       long sAm, long sAk, long sBk, long sBn, long sCm, long sCn,
       long aBo, long aBi, long bBo, long bBi, long cBo, long cBi,
       long rBo, long rBi, int nInner, int act)
{
  __shared__ alignas(16) _Float16 As[64][40];   // m-major
  __shared__ alignas(16) _Float16 Bs[64][40];   // n-major
  int z = blockIdx.z, zo = z / nInner, zi = z % nInner;
  const float* Ab = A + (long)zo * aBo + (long)zi * aBi;
  const float* Bb = B + (long)zo * bBo + (long)zi * bBi;
  float*       Cb = C + (long)zo * cBo + (long)zi * cBi;
  const float* Rb = res ? res + (long)zo * rBo + (long)zi * rBi : (const float*)0;
  int m0 = blockIdx.y * 64, n0 = blockIdx.x * 64;
  int t = threadIdx.x, lane = t & 31, w = t >> 5;
  int wr = (w >> 1) * 32, wc = (w & 1) * 32;

  // staging thread maps
  int a_m  = t >> 1, a_kb = (t & 1) * 16;   // k-contiguous map
  int a_kk = t >> 2, a_mb = (t & 3) * 16;   // m-contiguous map
  int b_n  = t >> 1, b_kb = (t & 1) * 16;   // k-contiguous map
  int b_kk = t >> 2, b_nb = (t & 3) * 16;   // n-contiguous map
  bool aFull = (m0 + 64 <= M);
  bool bFull = (n0 + 64 <= N);

  v8f acc00 = {}, acc01 = {}, acc10 = {}, acc11 = {};

  for (int k0 = 0; k0 < K; k0 += 32) {
    bool kf = (k0 + 32 <= K);
    if (aFull && kf) {
      if (sAk == 1) {
        const float* p = Ab + (long)(m0 + a_m) * sAm + (k0 + a_kb);
        const f32x4* q = (const f32x4*)p;
        cvt_store16(&As[a_m][a_kb], q[0], q[1], q[2], q[3]);
      } else if (sAm == 1) {
        const float* p = Ab + (long)(k0 + a_kk) * sAk + (m0 + a_mb);
        const f32x4* q = (const f32x4*)p;
        cvt_scatter16(&As[a_mb][a_kk], 40, q[0], q[1], q[2], q[3]);
      } else {
        const float* p = Ab + (long)(m0 + a_m) * sAm + (long)(k0 + a_kb) * sAk;
        ldstrided_store16(&As[a_m][a_kb], p, sAk);
      }
    } else {
      int gm = m0 + a_m, ka = k0 + a_kb;
      bool ok = (gm < M) && (K - ka >= 16);
      const float* p = Ab + (long)(gm < M ? gm : M - 1) * sAm
                          + (long)(ka < K ? ka : K - 1) * sAk;
      ldmask_store16(&As[a_m][a_kb], p, ok ? sAk : 0, ok);
    }
    if (bFull && kf) {
      if (sBk == 1) {
        const float* p = Bb + (long)(n0 + b_n) * sBn + (k0 + b_kb);
        const f32x4* q = (const f32x4*)p;
        cvt_store16(&Bs[b_n][b_kb], q[0], q[1], q[2], q[3]);
        if (k0 + 32 < K) __builtin_prefetch(p + 32, 0, 1);
      } else if (sBn == 1) {
        const float* p = Bb + (long)(k0 + b_kk) * sBk + (n0 + b_nb);
        const f32x4* q = (const f32x4*)p;
        cvt_scatter16(&Bs[b_nb][b_kk], 40, q[0], q[1], q[2], q[3]);
        if (k0 + 32 < K) __builtin_prefetch(p + 32 * sBk, 0, 1);
      } else {
        const float* p = Bb + (long)(n0 + b_n) * sBn + (long)(k0 + b_kb) * sBk;
        ldstrided_store16(&Bs[b_n][b_kb], p, sBk);
      }
    } else {
      int gn = n0 + b_n, kb = k0 + b_kb;
      bool ok = (gn < N) && (K - kb >= 16);
      const float* q = Bb + (long)(gn < N ? gn : N - 1) * sBn
                          + (long)(kb < K ? kb : K - 1) * sBk;
      ldmask_store16(&Bs[b_n][b_kb], q, ok ? sBk : 0, ok);
    }
    __syncthreads();
    v16h a0 = frag_a_ld(&As[wr][0], 40);
    v16h a1 = frag_a_ld(&As[wr + 16][0], 40);
    v16h b0 = frag_b_ld(&Bs[wc][0], 40);
    v16h b1 = frag_b_ld(&Bs[wc + 16][0], 40);
    acc00 = wmma16(a0, b0, acc00);
    acc01 = wmma16(a0, b1, acc01);
    acc10 = wmma16(a1, b0, acc10);
    acc11 = wmma16(a1, b1, acc11);
    __syncthreads();
  }

  int ln = lane & 15, g = lane >> 4;
#define STORE_FRAG(ACC, FR, FC)                                            \
  {                                                                        \
    _Pragma("unroll") for (int r = 0; r < 8; ++r) {                        \
      int m = m0 + wr + (FR)*16 + 8 * g + r;                               \
      int n = n0 + wc + (FC)*16 + ln;                                      \
      if (m < M && n < N) {                                                \
        float v = (ACC)[r];                                                \
        if (bias) v += bias[m];                                            \
        long off = (long)m * sCm + (long)n * sCn;                          \
        if (Rb) v += Rb[off];                                              \
        if (act) v = v >= 0.f ? v : 0.01f * v;                             \
        Cb[off] = v;                                                       \
      }                                                                    \
    }                                                                      \
  }
  STORE_FRAG(acc00, 0, 0)
  STORE_FRAG(acc01, 0, 1)
  STORE_FRAG(acc10, 1, 0)
  STORE_FRAG(acc11, 1, 1)
#undef STORE_FRAG
}

// ---------------------------------------------------------------------------
// Implicit-GEMM conv (NCHW). KHxKW taps, stride, zero or reflect pad.
// Block: 16 cout x 64 output pixels (one row segment), 4 waves. A full tap
// (all Cin channels) is staged per barrier pair -> Cin/32 WMMAs per sync,
// with the reflect/clamp index math done once per thread per tap.
// ---------------------------------------------------------------------------
__global__ void __launch_bounds__(128)
k_conv(const float* __restrict__ X, const float* __restrict__ Wt,
       const float* __restrict__ res, const float* __restrict__ preluA,
       float* __restrict__ Y,
       int Cin, int Cout, int Hin, int Win, int Hout, int Wout,
       int KH, int KW, int stride, int pad, int reflect,
       int outTotC, int outCOff, int resTotC)
{
  __shared__ alignas(16) _Float16 Wsh[16][CONV_LD];   // m-major (cout x ci)
  __shared__ alignas(16) _Float16 Xsh[64][CONV_LD];   // n-major (pixel x ci)
  int t = threadIdx.x, lane = t & 31, w = t >> 5;
  long HWo = (long)Hout * Wout;
  long planes = (long)Hin * Win;
  long n0 = (long)blockIdx.x * 64;
  int b  = (int)(n0 / HWo);
  int rp = (int)(n0 - (long)b * HWo);
  int oy = rp / Wout, ox0 = rp % Wout;
  int co0 = blockIdx.y * 16;
  const float* Xb = X + (long)b * Cin * planes;
  v8f acc = {};
  int taps = KH * KW;

  int pp  = t >> 1, ckb = (t & 1) * 16;   // input staging: one pixel, 16 channels
  int wm  = t >> 3, wk4 = (t & 7) * 4;    // weight staging: one cout row, 4 channels

  for (int tap = 0; tap < taps; ++tap) {
    int ky = tap / KW, kx = tap % KW;
    int iy = oy * stride - pad + ky;
    if (reflect) { iy = iy < 0 ? -iy : iy; iy = iy >= Hin ? 2 * Hin - 2 - iy : iy; }
    bool yok = (iy >= 0 && iy < Hin);
    int iyc = iy < 0 ? 0 : (iy >= Hin ? Hin - 1 : iy);
    int ix = (ox0 + pp) * stride - pad + kx;
    if (reflect) { ix = ix < 0 ? -ix : ix; ix = ix >= Win ? 2 * Win - 2 - ix : ix; }
    bool pok = yok && ix >= 0 && ix < Win;
    int ixc = ix < 0 ? 0 : (ix >= Win ? Win - 1 : ix);
    const float* px = Xb + (long)iyc * Win + ixc;

    // stage the whole tap: all Cin channels
    for (int kc = 0; kc < Cin; kc += 32) {
      int kcb = kc + ckb;
      bool cok = (Cin - kcb >= 16);           // all-or-nothing (Cin % 16 == 0)
      const float* p = px + (long)(kcb < Cin ? kcb : Cin - 1) * planes;
      ldmask_store16(&Xsh[pp][kcb], p, cok ? planes : 0, cok && pok);
#pragma unroll
      for (int i = 0; i < 4; ++i) {
        int ci = kc + wk4 + i;
        int cic = ci < Cin ? ci : Cin - 1;
        float v = Wt[(((long)(co0 + wm) * Cin + cic) * KH + ky) * KW + kx];
        Wsh[wm][kc + wk4 + i] = (_Float16)(ci < Cin ? v : 0.f);
      }
    }
    __syncthreads();
    for (int kc = 0; kc < Cin; kc += 32) {
      v16h a  = frag_a_ld(&Wsh[0][kc], CONV_LD);
      v16h bb = frag_b_ld(&Xsh[w * 16][kc], CONV_LD);
      acc = wmma16(a, bb, acc);
    }
    __syncthreads();
  }

  int ln = lane & 15, g = lane >> 4;
  float pa = preluA ? preluA[0] : 0.f;
#pragma unroll
  for (int r = 0; r < 8; ++r) {
    int co = co0 + 8 * g + r;
    int p  = rp + w * 16 + ln;
    float v = acc[r];
    if (preluA && v < 0.f) v *= pa;
    if (res) v += res[((long)b * resTotC + co) * HWo + p];
    Y[((long)b * outTotC + outCOff + co) * HWo + p] = v;
  }
}

// ---------------------------------------------------------------------------
// ConvTranspose2d(Cin, Cout, 4, stride=2, pad=1) gather form, full-tap staged.
// ---------------------------------------------------------------------------
__global__ void __launch_bounds__(128)
k_convt(const float* __restrict__ X, const float* __restrict__ Wt,
        const float* __restrict__ bias, float* __restrict__ Y,
        int Cin, int Cout, int Hin, int Win, int Hout, int Wout,
        int outTotC, int outCOff)
{
  __shared__ alignas(16) _Float16 Wsh[16][CONV_LD];
  __shared__ alignas(16) _Float16 Xsh[64][CONV_LD];
  int t = threadIdx.x, lane = t & 31, w = t >> 5;
  long HWo = (long)Hout * Wout;
  long planes = (long)Hin * Win;
  long n0 = (long)blockIdx.x * 64;
  int b  = (int)(n0 / HWo);
  int rp = (int)(n0 - (long)b * HWo);
  int oy = rp / Wout, ox0 = rp % Wout;
  int co0 = blockIdx.y * 16;
  const float* Xb = X + (long)b * Cin * planes;
  v8f acc = {};

  int pp  = t >> 1, ckb = (t & 1) * 16;
  int wm  = t >> 3, wk4 = (t & 7) * 4;

  for (int tap = 0; tap < 16; ++tap) {
    int ky = tap >> 2, kx = tap & 3;
    int ty = oy + 1 - ky;
    bool yok = (ty >= 0) && ((ty & 1) == 0) && ((ty >> 1) < Hin);
    int iy = ty >> 1;
    int iyc = iy < 0 ? 0 : (iy >= Hin ? Hin - 1 : iy);
    int tx = (ox0 + pp) + 1 - kx;
    bool pok = yok && tx >= 0 && ((tx & 1) == 0) && ((tx >> 1) < Win);
    int ix = tx >> 1;
    int ixc = ix < 0 ? 0 : (ix >= Win ? Win - 1 : ix);
    const float* px = Xb + (long)iyc * Win + ixc;

    for (int kc = 0; kc < Cin; kc += 32) {
      int kcb = kc + ckb;
      bool cok = (Cin - kcb >= 16);
      const float* p = px + (long)(kcb < Cin ? kcb : Cin - 1) * planes;
      ldmask_store16(&Xsh[pp][kcb], p, cok ? planes : 0, cok && pok);
#pragma unroll
      for (int i = 0; i < 4; ++i) {
        int ci = kc + wk4 + i;
        int cic = ci < Cin ? ci : Cin - 1;
        float v = Wt[((long)cic * Cout + (co0 + wm)) * 16 + tap];
        Wsh[wm][kc + wk4 + i] = (_Float16)(ci < Cin ? v : 0.f);
      }
    }
    __syncthreads();
    for (int kc = 0; kc < Cin; kc += 32) {
      v16h a  = frag_a_ld(&Wsh[0][kc], CONV_LD);
      v16h bb = frag_b_ld(&Xsh[w * 16][kc], CONV_LD);
      acc = wmma16(a, bb, acc);
    }
    __syncthreads();
  }

  int ln = lane & 15, g = lane >> 4;
#pragma unroll
  for (int r = 0; r < 8; ++r) {
    int co = co0 + 8 * g + r;
    int p  = rp + w * 16 + ln;
    float v = acc[r] + (bias ? bias[co] : 0.f);
    Y[((long)b * outTotC + outCOff + co) * HWo + p] = v;
  }
}

// ------------------------------ VALU glue ----------------------------------

__global__ void k_add3(const float* a, const float* b, const float* c,
                       float* o, long n) {
  long i = (long)blockIdx.x * 256 + threadIdx.x, st = (long)gridDim.x * 256;
  for (; i < n; i += st) o[i] = a[i] + b[i] + c[i];
}

__global__ void k_chan_scale(float* x, const float* s, long hw, long n) {
  long i = (long)blockIdx.x * 256 + threadIdx.x, st = (long)gridDim.x * 256;
  for (; i < n; i += st) x[i] *= s[i / hw];
}

__global__ void k_head_scale(float* x, const float* rs, long n) {
  long i = (long)blockIdx.x * 256 + threadIdx.x, st = (long)gridDim.x * 256;
  for (; i < n; i += st) x[i] *= rs[(i >> 12) % 3];   // 64*64 = 4096 per head
}

__global__ void k_softmax_rows(float* x, int L) {
  __shared__ float red[256];
  long row = blockIdx.x;
  float* p = x + row * (long)L;
  int t = threadIdx.x;
  float m = -3.4e38f;
  for (int i = t; i < L; i += 256) m = fmaxf(m, p[i]);
  red[t] = m; __syncthreads();
  for (int s = 128; s > 0; s >>= 1) { if (t < s) red[t] = fmaxf(red[t], red[t + s]); __syncthreads(); }
  float mx = red[0]; __syncthreads();
  float sum = 0.f;
  for (int i = t; i < L; i += 256) { float e = __expf(p[i] - mx); p[i] = e; sum += e; }
  red[t] = sum; __syncthreads();
  for (int s = 128; s > 0; s >>= 1) { if (t < s) red[t] += red[t + s]; __syncthreads(); }
  float inv = 1.f / red[0];
  for (int i = t; i < L; i += 256) p[i] *= inv;
}

// aug[b,c] = sum_n g[b,c,n] * a[b,n]
__global__ void k_rowdot(const float* g, const float* a, float* o, int Cc, int N) {
  __shared__ float red[256];
  long row = blockIdx.x;
  long b = row / Cc;
  const float* gp = g + row * (long)N;
  const float* ap = a + b * (long)N;
  int t = threadIdx.x;
  float s = 0.f;
  for (int i = t; i < N; i += 256) s += gp[i] * ap[i];
  red[t] = s; __syncthreads();
  for (int st = 128; st > 0; st >>= 1) { if (t < st) red[t] += red[t + st]; __syncthreads(); }
  if (t == 0) o[row] = red[0];
}

__global__ void k_layernorm(const float* x, const float* w, const float* b,
                            float* y, int D) {
  __shared__ float red[64];
  long row = blockIdx.x;
  const float* p = x + row * (long)D;
  float* q = y + row * (long)D;
  int t = threadIdx.x;
  float s = 0.f;
  for (int i = t; i < D; i += 64) s += p[i];
  red[t] = s; __syncthreads();
  for (int st = 32; st > 0; st >>= 1) { if (t < st) red[t] += red[t + st]; __syncthreads(); }
  float mu = red[0] / D; __syncthreads();
  float v = 0.f;
  for (int i = t; i < D; i += 64) { float d = p[i] - mu; v += d * d; }
  red[t] = v; __syncthreads();
  for (int st = 32; st > 0; st >>= 1) { if (t < st) red[t] += red[t + st]; __syncthreads(); }
  float inv = rsqrtf(red[0] / D + 1e-5f);
  for (int i = t; i < D; i += 64) q[i] = (p[i] - mu) * inv * w[i] + b[i];
}

// L2-normalize over n for x laid out (b, n, Cc): one block per (b, ch)
__global__ void k_l2norm_bnc(float* x, int Cc, int N) {
  __shared__ float red[256];
  long row = blockIdx.x;
  long b = row / Cc; int c = (int)(row % Cc);
  float* base = x + b * (long)N * Cc + c;
  int t = threadIdx.x;
  float s = 0.f;
  for (int i = t; i < N; i += 256) { float v = base[(long)i * Cc]; s += v * v; }
  red[t] = s; __syncthreads();
  for (int st = 128; st > 0; st >>= 1) { if (t < st) red[t] += red[t + st]; __syncthreads(); }
  float inv = 1.f / fmaxf(sqrtf(red[0]), 1e-12f);
  for (int i = t; i < N; i += 256) base[(long)i * Cc] *= inv;
}

__global__ void k_nchw2nhwc(const float* x, float* y, int Cc, int N, long total) {
  long i = (long)blockIdx.x * 256 + threadIdx.x, st = (long)gridDim.x * 256;
  for (; i < total; i += st) {
    long b = i / ((long)Cc * N); long r = i - b * (long)Cc * N;
    int c = (int)(r / N); int n = (int)(r - (long)c * N);
    y[(b * N + n) * Cc + c] = x[i];
  }
}

__global__ void k_nhwc2nchw(const float* x, float* y, int Cc, int N, long total) {
  long i = (long)blockIdx.x * 256 + threadIdx.x, st = (long)gridDim.x * 256;
  for (; i < total; i += st) {
    long b = i / ((long)Cc * N); long r = i - b * (long)Cc * N;
    int c = (int)(r / N); int n = (int)(r - (long)c * N);
    y[i] = x[(b * N + n) * Cc + c];
  }
}

// ------------------------------ host side ----------------------------------

static inline unsigned blocks_for(long n) {
  long b = (n + 255) / 256;
  if (b > 65535) b = 65535;
  return (unsigned)b;
}

static void gemm(hipStream_t s, const float* A, const float* B, const float* bias,
                 const float* res, float* C, int M, int N, int K,
                 long sAm, long sAk, long sBk, long sBn, long sCm, long sCn,
                 long aBo, long aBi, long bBo, long bBi, long cBo, long cBi,
                 long rBo, long rBi, int nOuter, int nInner, int act)
{
  dim3 g((N + 63) / 64, (M + 63) / 64, nOuter * nInner);
  k_gemm<<<g, 128, 0, s>>>(A, B, bias, res, C, M, N, K, sAm, sAk, sBk, sBn,
                           sCm, sCn, aBo, aBi, bBo, bBi, cBo, cBi, rBo, rBi,
                           nInner, act);
}

static void conv(hipStream_t s, const float* X, const float* Wt, const float* res,
                 const float* pa, float* Y, int B, int Cin, int Cout,
                 int Hin, int Win, int Hout, int Wout, int KH, int KW,
                 int stride, int pad, int reflect,
                 int outTotC, int outCOff, int resTotC)
{
  long NPo = (long)B * Hout * Wout;
  dim3 g((unsigned)(NPo / 64), Cout / 16, 1);
  k_conv<<<g, 128, 0, s>>>(X, Wt, res, pa, Y, Cin, Cout, Hin, Win, Hout, Wout,
                           KH, KW, stride, pad, reflect, outTotC, outCOff, resTotC);
}

extern "C" void kernel_launch(void* const* d_in, const int* in_sizes, int n_in,
                              void* d_out, int out_size, void* d_ws, size_t ws_size,
                              hipStream_t stream)
{
  const float* in[64];
  for (int i = 0; i < n_in && i < 64; ++i) in[i] = (const float*)d_in[i];
  float* out = (float*)d_out;
  float* ws  = (float*)d_ws;

  const int  Bn = 8, Hh = 128, Ww = 128, H2 = 64;
  const long HW = (long)Hh * Ww;          // 16384
  const long HW2 = (long)H2 * H2;         // 4096
  const long SZ96  = (long)Bn * 96  * HW;
  const long SZ48  = (long)Bn * 48  * HW;
  const long SZ32  = (long)Bn * 32  * HW;
  const long SZ144 = (long)Bn * 144 * HW;
  const long SZS   = (long)Bn * 192 * HW;
  const long SZT   = (long)Bn * 192 * HW2;

  long off = 0;
  float* OUT  = ws + off; off += SZ96;
  float* TA   = ws + off; off += SZ96;
  float* TB   = ws + off; off += SZ96;
  float* OUT1 = ws + off; off += SZ96;
  float* X0   = ws + off; off += SZ96;
  float* X1   = ws + off; off += SZ48;
  float* X2   = ws + off; off += SZ32;
  float* FALL = ws + off; off += SZ144;
  float* Sb   = ws + off; off += SZS;
  float* XT   = ws + off; off += SZT;
  float* XU   = ws + off; off += SZT;
  float* LNB  = ws + off; off += SZT;
  float* QB   = ws + off; off += SZT;
  float* KB   = ws + off; off += SZT;
  float* VB   = ws + off; off += SZT;
  float* OB   = ws + off; off += SZT;
  float* ATT  = ws + off; off += (long)Bn * 3 * 64 * 64;
  float* M1b  = ws + off; off += (long)Bn * 96 * 48;
  float* M2b  = ws + off; off += (long)Bn * 48 * 32;
  float* LOG  = ws + off; off += (long)Bn * HW;
  float* AUG  = ws + off; off += (long)Bn * 96;
  float* LOG2 = ws + off; off += (long)Bn * HW2;
  float* AUG2 = ws + off; off += (long)Bn * 192;
  (void)ws_size; (void)in_sizes; (void)out_size;

  const float* x = in[0];

  // ===== LCAB =====
  conv(stream, x,  in[1], 0, in[2], TA,  Bn, 96, 96, Hh, Ww, Hh, Ww, 3,3,1,1,1, 96,0,0);
  conv(stream, TA, in[3], 0, 0,     OUT, Bn, 96, 96, Hh, Ww, Hh, Ww, 3,3,1,1,1, 96,0,0);
  conv(stream, OUT, in[19], 0, in[20], TB, Bn, 96, 96, Hh, Ww, Hh, Ww, 3,3,1,1,1, 96,0,0);
  conv(stream, TB,  in[21], 0, 0,      TA, Bn, 96, 96, Hh, Ww, Hh, Ww, 3,3,1,1,1, 96,0,0);
  gemm(stream, in[22], TA, 0, 0, OUT1, 96, (int)HW, 96,
       96,1, HW,1, HW,1, 0,0, 96*HW,0, 96*HW,0, 0,0, Bn,1, 0);
  conv(stream, OUT, in[4], 0, in[5], TB, Bn, 96, 96, Hh, Ww, Hh, Ww, 3,3,1,1,1, 96,0,0);
  conv(stream, TB,  in[6], 0, 0,     TA, Bn, 96, 96, Hh, Ww, Hh, Ww, 3,3,1,1,1, 96,0,0);
  gemm(stream, in[7], TA, 0, 0, X0, 96, (int)HW, 96,
       96,1, HW,1, HW,1, 0,0, 96*HW,0, 96*HW,0, 0,0, Bn,1, 0);
  conv(stream, OUT, in[8],  0, in[9], TB, Bn, 96, 48, Hh, Ww, Hh, Ww, 3,3,1,1,1, 48,0,0);
  conv(stream, TB,  in[10], 0, 0,     TA, Bn, 48, 48, Hh, Ww, Hh, Ww, 3,3,1,1,1, 48,0,0);
  gemm(stream, in[11], TA, 0, 0, X1, 48, (int)HW, 48,
       48,1, HW,1, HW,1, 0,0, 48*HW,0, 48*HW,0, 0,0, Bn,1, 0);
  conv(stream, OUT, in[12], 0, in[13], TB, Bn, 96, 32, Hh, Ww, Hh, Ww, 3,3,1,1,1, 32,0,0);
  conv(stream, TB,  in[14], 0, 0,      TA, Bn, 32, 32, Hh, Ww, Hh, Ww, 3,3,1,1,1, 32,0,0);
  gemm(stream, in[15], TA, 0, 0, X2, 32, (int)HW, 32,
       32,1, HW,1, HW,1, 0,0, 32*HW,0, 32*HW,0, 0,0, Bn,1, 0);
  gemm(stream, X0, X1, 0, 0, M1b, 96, 48, (int)HW,
       HW,1, 1,HW, 48,1, 96*HW,0, 48*HW,0, 96*48,0, 0,0, Bn,1, 0);
  gemm(stream, M1b, X1, 0, X0, FALL, 96, (int)HW, 48,
       48,1, HW,1, 1,96, 96*48,0, 48*HW,0, 144*HW,0, 96*HW,0, Bn,1, 0);
  gemm(stream, X1, X2, 0, 0, M2b, 48, 32, (int)HW,
       HW,1, 1,HW, 32,1, 48*HW,0, 32*HW,0, 48*32,0, 0,0, Bn,1, 0);
  gemm(stream, M2b, X2, 0, X1, FALL + 96*HW, 48, (int)HW, 32,
       32,1, HW,1, 1,48, 48*32,0, 32*HW,0, 144*HW,0, 48*HW,0, Bn,1, 0);
  conv(stream, FALL, in[16], 0,   in[17], TB, Bn, 144, 96, Hh, Ww, Hh, Ww, 3,3,1,1,1, 96,0,0);
  conv(stream, TB,   in[18], OUT, 0,      TA, Bn, 96,  96, Hh, Ww, Hh, Ww, 3,3,1,1,1, 96,0,96);
  k_add3<<<blocks_for(SZ96), 256, 0, stream>>>(OUT1, TA, x, TB, SZ96);
  conv(stream, TB, in[23], 0, in[24], TA, Bn, 96, 96, Hh, Ww, Hh, Ww, 3,3,1,1,1, 96,0,0);
  conv(stream, TA, in[25], x, 0,      Sb, Bn, 96, 96, Hh, Ww, Hh, Ww, 3,3,1,1,1, 192,96,96);

  // ===== GLAF0 (downsample) =====
  gemm(stream, in[26], x, 0, 0, TA, 96, (int)HW, 96,
       96,1, HW,1, HW,1, 0,0, 96*HW,0, 96*HW,0, 0,0, Bn,1, 0);
  gemm(stream, in[27], TA, 0, 0, LOG, 1, (int)HW, 96,
       96,1, HW,1, HW,1, 0,0, 96*HW,0, HW,0, 0,0, Bn,1, 0);
  k_softmax_rows<<<Bn, 256, 0, stream>>>(LOG, (int)HW);
  k_rowdot<<<Bn * 96, 256, 0, stream>>>(TA, LOG, AUG, 96, (int)HW);
  k_chan_scale<<<blocks_for(SZ96), 256, 0, stream>>>(TA, AUG, HW, SZ96);
  conv(stream, TA, in[28], 0, 0, XU, Bn, 96, 192, Hh, Ww, H2, H2, 4,4,2,1,0, 192,0,0);
  k_nchw2nhwc<<<blocks_for(SZT), 256, 0, stream>>>(XU, XT, 192, (int)HW2, SZT);

  // ===== Transformer (2 layers), tokens (b, 4096, 192) in XT =====
  for (int l = 0; l < 2; ++l) {
    int L = 29 + 14 * l;
    const long TB_ = HW2 * 192;
    k_layernorm<<<(unsigned)(Bn * HW2), 64, 0, stream>>>(XT, in[L], in[L+1], LNB, 192);
    gemm(stream, in[L+2], LNB, 0, 0, QB, 192, (int)HW2, 192,
         192,1, 1,192, 1,192, 0,0, TB_,0, TB_,0, 0,0, Bn,1, 0);
    gemm(stream, in[L+3], LNB, 0, 0, KB, 192, (int)HW2, 192,
         192,1, 1,192, 1,192, 0,0, TB_,0, TB_,0, 0,0, Bn,1, 0);
    gemm(stream, in[L+4], LNB, 0, 0, VB, 192, (int)HW2, 192,
         192,1, 1,192, 1,192, 0,0, TB_,0, TB_,0, 0,0, Bn,1, 0);
    k_l2norm_bnc<<<Bn * 192, 256, 0, stream>>>(QB, 192, (int)HW2);
    k_l2norm_bnc<<<Bn * 192, 256, 0, stream>>>(KB, 192, (int)HW2);
    gemm(stream, KB, QB, 0, 0, ATT, 64, 64, (int)HW2,
         1,192, 192,1, 64,1, TB_,64, TB_,64, 3*64*64,64*64, 0,0, Bn,3, 0);
    k_head_scale<<<blocks_for((long)Bn*3*64*64), 256, 0, stream>>>(ATT, in[L+5], (long)Bn*3*64*64);
    k_softmax_rows<<<Bn * 3 * 64, 256, 0, stream>>>(ATT, 64);
    gemm(stream, ATT, VB, 0, 0, OB, 64, (int)HW2, 64,
         64,1, 1,192, 1,192, 3*64*64,64*64, TB_,64, TB_,64, 0,0, Bn,3, 0);
    gemm(stream, in[L+6], OB, in[L+7], XT, XU, 192, (int)HW2, 192,
         192,1, 1,192, 1,192, 0,0, TB_,0, TB_,0, TB_,0, Bn,1, 0);
    k_layernorm<<<(unsigned)(Bn * HW2), 64, 0, stream>>>(XU, in[L+8], in[L+9], LNB, 192);
    gemm(stream, in[L+10], LNB, in[L+11], 0, QB, 192, (int)HW2, 192,
         192,1, 1,192, 1,192, 0,0, TB_,0, TB_,0, 0,0, Bn,1, 1);
    gemm(stream, in[L+12], QB, in[L+13], XU, XT, 192, (int)HW2, 192,
         192,1, 1,192, 1,192, 0,0, TB_,0, TB_,0, TB_,0, Bn,1, 0);
  }
  k_nhwc2nchw<<<blocks_for(SZT), 256, 0, stream>>>(XT, XU, 192, (int)HW2, SZT);

  // ===== GLAF1 (upsample) =====
  gemm(stream, in[57], XU, 0, 0, LNB, 192, (int)HW2, 192,
       192,1, HW2,1, HW2,1, 0,0, 192*HW2,0, 192*HW2,0, 0,0, Bn,1, 0);
  gemm(stream, in[58], LNB, 0, 0, LOG2, 1, (int)HW2, 192,
       192,1, HW2,1, HW2,1, 0,0, 192*HW2,0, HW2,0, 0,0, Bn,1, 0);
  k_softmax_rows<<<Bn, 256, 0, stream>>>(LOG2, (int)HW2);
  k_rowdot<<<Bn * 192, 256, 0, stream>>>(LNB, LOG2, AUG2, 192, (int)HW2);
  k_chan_scale<<<blocks_for(SZT), 256, 0, stream>>>(LNB, AUG2, HW2, SZT);
  {
    long NPo = (long)Bn * Hh * Ww;
    dim3 g((unsigned)(NPo / 64), 96 / 16, 1);
    k_convt<<<g, 128, 0, stream>>>(LNB, in[59], in[60], Sb,
                                   192, 96, H2, H2, Hh, Ww, 192, 0);
  }

  // ===== final reshape_blk(192->96) + x =====
  conv(stream, Sb, in[61], 0, in[62], TA,  Bn, 192, 96, Hh, Ww, Hh, Ww, 3,3,1,1,1, 96,0,0);
  conv(stream, TA, in[63], x, 0,      out, Bn, 96,  96, Hh, Ww, Hh, Ww, 3,3,1,1,1, 96,0,96);
}
